// MambaTTSDecoder_65017214927440
// MI455X (gfx1250) — compile-verified, matrix operand order checked by reading, not run on hardware
//
#include <hip/hip_runtime.h>
#include <hip/hip_bf16.h>
#include <math.h>

// ---------------------------------------------------------------------------
// Model constants (match the JAX reference)
// ---------------------------------------------------------------------------
#define D_MODEL 512
#define D_INNER 1024
#define D_STATE 16
#define DT_RANK 32
#define D_CONV  4
#define N_HEADS 8
#define D_FF    2048
#define D_STYLE 256
#define N_LAYERS 8
#define VOCAB   1024
#define BB      2
#define TT      1024
#define T_TEXT  256
#define ROWS    (BB * TT)           // 2048 token rows
#define KV_ROWS (BB * T_TEXT)       // 512 text rows
#define DH      (D_MODEL / N_HEADS) // 64

typedef __attribute__((ext_vector_type(16))) __bf16 v16bf;
typedef __attribute__((ext_vector_type(8)))  float  v8f;
typedef __attribute__((ext_vector_type(4)))  float  f4;

#if __has_builtin(__builtin_amdgcn_sched_barrier)
#define SCHED_FENCE() __builtin_amdgcn_sched_barrier(0)
#else
#define SCHED_FENCE()
#endif

// ---------------------------------------------------------------------------
// Wave32 reductions
// ---------------------------------------------------------------------------
__device__ __forceinline__ float waveMax(float v) {
#pragma unroll
  for (int m = 16; m >= 1; m >>= 1) v = fmaxf(v, __shfl_xor(v, m, 32));
  return v;
}
__device__ __forceinline__ float waveSum(float v) {
#pragma unroll
  for (int m = 16; m >= 1; m >>= 1) v += __shfl_xor(v, m, 32);
  return v;
}

// ---------------------------------------------------------------------------
// 1) Embedding: x = tok_emb[tok] + pos_emb[t] + quant_emb
// ---------------------------------------------------------------------------
__global__ __launch_bounds__(256) void embed_kernel(
    const int* __restrict__ tok, const float* __restrict__ tok_emb,
    const float* __restrict__ pos_emb, const float* __restrict__ quant_emb,
    float* __restrict__ X) {
  int r = blockIdx.x;                // token row 0..2047
  int t = r % TT;
  int tk = tok[r];
  for (int d = threadIdx.x; d < D_MODEL; d += 256) {
    X[(size_t)r * D_MODEL + d] =
        tok_emb[(size_t)tk * D_MODEL + d] + pos_emb[(size_t)t * D_MODEL + d] +
        quant_emb[d];
  }
}

// ---------------------------------------------------------------------------
// 2) LayerNorm over 512 features: one wave per row (16 elems / lane)
// ---------------------------------------------------------------------------
__global__ __launch_bounds__(256) void ln_kernel(
    const float* __restrict__ X, const float* __restrict__ w,
    const float* __restrict__ b, float* __restrict__ Y, int rows) {
  int lane = threadIdx.x & 31;
  int row = blockIdx.x * 8 + (threadIdx.x >> 5);
  if (row >= rows) return;
  const float* xr = X + (size_t)row * D_MODEL;
  float v[16];
  float s = 0.f;
#pragma unroll
  for (int j = 0; j < 16; ++j) { v[j] = xr[j * 32 + lane]; s += v[j]; }
  float mu = waveSum(s) * (1.0f / D_MODEL);
  float s2 = 0.f;
#pragma unroll
  for (int j = 0; j < 16; ++j) { float c = v[j] - mu; s2 += c * c; }
  float var = waveSum(s2) * (1.0f / D_MODEL);
  float inv = rsqrtf(var + 1e-5f);
  float* yr = Y + (size_t)row * D_MODEL;
#pragma unroll
  for (int j = 0; j < 16; ++j) {
    int c = j * 32 + lane;
    yr[c] = (v[j] - mu) * inv * w[c] + b[c];
  }
}

// ---------------------------------------------------------------------------
// Fragment helpers.  Per CDNA5 ISA 7.12.2, lane (row=lane%16, hf=lane/16)
// holds bf16 K-pairs at K = hf*8 + {0..7} and 16 + hf*8 + {0..7}.  With hf*8
// folded into the row pointer these are two contiguous 8-float runs:
// four b128 loads, then packed f32->bf16 converts.
// ---------------------------------------------------------------------------
struct RawFrag { f4 x0, x1, x2, x3; };

__device__ __forceinline__ RawFrag load_raw(const float* __restrict__ p) {
  RawFrag r;
  r.x0 = *(const f4*)(p);
  r.x1 = *(const f4*)(p + 4);
  r.x2 = *(const f4*)(p + 16);
  r.x3 = *(const f4*)(p + 20);
  return r;
}

__device__ __forceinline__ v16bf cvt_frag(const RawFrag& f) {
  v16bf r;
  r[0]  = (__bf16)f.x0.x; r[1]  = (__bf16)f.x0.y;
  r[2]  = (__bf16)f.x0.z; r[3]  = (__bf16)f.x0.w;
  r[4]  = (__bf16)f.x1.x; r[5]  = (__bf16)f.x1.y;
  r[6]  = (__bf16)f.x1.z; r[7]  = (__bf16)f.x1.w;
  r[8]  = (__bf16)f.x2.x; r[9]  = (__bf16)f.x2.y;
  r[10] = (__bf16)f.x2.z; r[11] = (__bf16)f.x2.w;
  r[12] = (__bf16)f.x3.x; r[13] = (__bf16)f.x3.y;
  r[14] = (__bf16)f.x3.z; r[15] = (__bf16)f.x3.w;
  return r;
}

// ---------------------------------------------------------------------------
// 3) GEMM: Y[M,N] = act(A[M,K(lda)] @ W[N,K]^T + bias) + R
//    bf16 WMMA, f32 accumulate. One wave computes a 16x64 tile (4 WMMAs/K-step).
//    sched_barriers force: [20 b128 loads] | [40 pk-cvts] | [4 WMMAs], so the
//    loads clause up with amortized s_wait_loadcnt and the WMMA B operands are
//    not recycled by the next loads (no WAR hazard v_nops).
//    REQUIRES: M % 16 == 0, N % 64 == 0, K % 32 == 0, lda % 16 == 0.
//    act: 0=none, 1=gelu(exact erf), 2=tanh
// ---------------------------------------------------------------------------
__global__ __launch_bounds__(256) void gemm_bf16_kernel(
    const float* __restrict__ A, int lda, const float* __restrict__ W,
    const float* __restrict__ bias, const float* __restrict__ R,
    float* __restrict__ Y, int M, int N, int K, int act) {
  const int lane = threadIdx.x & 31;
  const int wib  = threadIdx.x >> 5;
  const int tilesM = M >> 4;
  const int tilesN = N >> 6;
  const int wave = blockIdx.x * 8 + wib;
  if (wave >= tilesM * tilesN) return;
  const int mt = wave % tilesM;
  const int nt = wave / tilesM;
  const int ml = lane & 15;
  const int hf = lane >> 4;

  const float* Arow = A + (size_t)(mt * 16 + ml) * lda + hf * 8;
  const float* W0 = W + (size_t)(nt * 64 + ml) * K + hf * 8;
  const float* W1 = W0 + (size_t)16 * K;
  const float* W2 = W0 + (size_t)32 * K;
  const float* W3 = W0 + (size_t)48 * K;

  v8f acc0, acc1, acc2, acc3;
#pragma unroll
  for (int j = 0; j < 8; ++j) { acc0[j] = 0.f; acc1[j] = 0.f; acc2[j] = 0.f; acc3[j] = 0.f; }

  for (int k0 = 0; k0 < K; k0 += 32) {
    if (k0 + 32 < K) {
      __builtin_prefetch(Arow + k0 + 32, 0, 1);     // global_prefetch_b8
      __builtin_prefetch(W0 + k0 + 32, 0, 1);
    }
    // Phase 1: all raw loads; fence keeps every destination live at once so
    // RA assigns distinct regs and the loads issue as a clause.
    RawFrag ra = load_raw(Arow + k0);
    RawFrag r0 = load_raw(W0 + k0);
    RawFrag r1 = load_raw(W1 + k0);
    RawFrag r2 = load_raw(W2 + k0);
    RawFrag r3 = load_raw(W3 + k0);
    SCHED_FENCE();
    // Phase 2: convert
    v16bf af = cvt_frag(ra);
    v16bf b0 = cvt_frag(r0);
    v16bf b1 = cvt_frag(r1);
    v16bf b2 = cvt_frag(r2);
    v16bf b3 = cvt_frag(r3);
    SCHED_FENCE();
    // Phase 3: 4 WMMAs, distinct accumulators, shared A
    acc0 = __builtin_amdgcn_wmma_f32_16x16x32_bf16(false, af, false, b0,
                                                   (short)0, acc0, false, false);
    acc1 = __builtin_amdgcn_wmma_f32_16x16x32_bf16(false, af, false, b1,
                                                   (short)0, acc1, false, false);
    acc2 = __builtin_amdgcn_wmma_f32_16x16x32_bf16(false, af, false, b2,
                                                   (short)0, acc2, false, false);
    acc3 = __builtin_amdgcn_wmma_f32_16x16x32_bf16(false, af, false, b3,
                                                   (short)0, acc3, false, false);
  }

  // Epilogue: C/D layout -> lane holds col = lane%16, rows v + 8*hf
#pragma unroll
  for (int ns = 0; ns < 4; ++ns) {
    v8f accv = (ns == 0) ? acc0 : (ns == 1) ? acc1 : (ns == 2) ? acc2 : acc3;
    int c = nt * 64 + ns * 16 + ml;
    float bval = bias ? bias[c] : 0.0f;
#pragma unroll
    for (int v = 0; v < 8; ++v) {
      int r = mt * 16 + v + 8 * hf;
      float val = accv[v] + bval;
      if (act == 1) val = 0.5f * val * (1.0f + erff(val * 0.70710678118f));
      else if (act == 2) val = tanhf(val);
      if (R) val += R[(size_t)r * N + c];
      Y[(size_t)r * N + c] = val;
    }
  }
}

// ---------------------------------------------------------------------------
// 4) Causal depthwise conv (width 4) + SiLU.  xi = xz[:, 0:1024]
// ---------------------------------------------------------------------------
__global__ __launch_bounds__(256) void conv_silu_kernel(
    const float* __restrict__ XZ, const float* __restrict__ cw,
    const float* __restrict__ cb, float* __restrict__ OUT) {
  int idx = blockIdx.x * 256 + threadIdx.x;       // over 2048*1024
  int i = idx & (D_INNER - 1);
  int r = idx >> 10;
  int b = r >> 10;
  int t = r & (TT - 1);
  float s = cb[i];
#pragma unroll
  for (int k = 0; k < D_CONV; ++k) {
    int tt = t - (D_CONV - 1) + k;
    if (tt >= 0)
      s += XZ[((size_t)(b * TT + tt)) * (2 * D_INNER) + i] * cw[i * D_CONV + k];
  }
  OUT[(size_t)r * D_INNER + i] = s / (1.0f + __expf(-s));
}

// ---------------------------------------------------------------------------
// 5) Selective-state scan: one thread per (b, channel), 16 states in regs
// ---------------------------------------------------------------------------
__global__ __launch_bounds__(256) void scan_kernel(
    const float* __restrict__ DTP, const float* __restrict__ XI,
    const float* __restrict__ DBC, const float* __restrict__ A_log,
    const float* __restrict__ Dp, float* __restrict__ Yout) {
  int g = blockIdx.x * 256 + threadIdx.x;          // 0..2047
  int b = g >> 10;
  int i = g & (D_INNER - 1);
  float As[D_STATE], h[D_STATE];
#pragma unroll
  for (int s = 0; s < D_STATE; ++s) {
    As[s] = -__expf(A_log[(size_t)i * D_STATE + s]);
    h[s] = 0.0f;
  }
  float dpi = Dp[i];
  for (int t = 0; t < TT; ++t) {
    size_t r = (size_t)(b * TT + t);
    float x = DTP[r * D_INNER + i];
    float dt = (x > 20.0f) ? x : log1pf(__expf(x));
    float xi = XI[r * D_INNER + i];
    const float* bc = DBC + r * (DT_RANK + 2 * D_STATE);
    float y = dpi * xi;
#pragma unroll
    for (int s = 0; s < D_STATE; ++s) {
      float dA = __expf(dt * As[s]);
      h[s] = dA * h[s] + dt * bc[DT_RANK + s] * xi;
      y += h[s] * bc[DT_RANK + D_STATE + s];
    }
    Yout[r * D_INNER + i] = y;
  }
}

// ---------------------------------------------------------------------------
// 6) Gating: g = y * silu(z), z = xz[:, 1024:2048]
// ---------------------------------------------------------------------------
__global__ __launch_bounds__(256) void gate_kernel(
    const float* __restrict__ Yssm, const float* __restrict__ XZ,
    float* __restrict__ G) {
  int idx = blockIdx.x * 256 + threadIdx.x;
  int i = idx & (D_INNER - 1);
  int r = idx >> 10;
  float z = XZ[(size_t)r * (2 * D_INNER) + D_INNER + i];
  G[(size_t)r * D_INNER + i] =
      Yssm[(size_t)r * D_INNER + i] * (z / (1.0f + __expf(-z)));
}

// ---------------------------------------------------------------------------
// 7) Cross-attention core: one wave per (b, head, q) row, Tk = 256
// ---------------------------------------------------------------------------
__global__ __launch_bounds__(256) void attn_kernel(
    const float* __restrict__ Q, const float* __restrict__ Kt,
    const float* __restrict__ Vt, const int* __restrict__ mask,
    float* __restrict__ AO) {
  __shared__ float probs[8 * T_TEXT];
  int lane = threadIdx.x & 31;
  int wib  = threadIdx.x >> 5;
  int wid = blockIdx.x * 8 + wib;                  // 0..16383
  int b = wid >> 13;
  int rem = wid & 8191;
  int h = rem >> 10;
  int q = rem & (TT - 1);
  const float* qr = Q + ((size_t)(b * TT + q)) * D_MODEL + h * DH;

  float s[8];
#pragma unroll
  for (int j = 0; j < 8; ++j) {
    int k = j * 32 + lane;
    const float* kr = Kt + ((size_t)(b * T_TEXT + k)) * D_MODEL + h * DH;
    float d = 0.0f;
    for (int e = 0; e < DH; ++e) d += qr[e] * kr[e];
    d *= 0.125f;                                    // 1/sqrt(64)
    if (mask[b * T_TEXT + k] == 0) d = -1.0e9f;
    s[j] = d;
  }
  float mx = -3.4e38f;
#pragma unroll
  for (int j = 0; j < 8; ++j) mx = fmaxf(mx, s[j]);
  mx = waveMax(mx);
  float sum = 0.0f;
#pragma unroll
  for (int j = 0; j < 8; ++j) { s[j] = __expf(s[j] - mx); sum += s[j]; }
  sum = waveSum(sum);
  float inv = 1.0f / sum;
#pragma unroll
  for (int j = 0; j < 8; ++j) probs[wib * T_TEXT + j * 32 + lane] = s[j] * inv;
  __syncthreads();

  int d0 = lane * 2;
  float a0 = 0.0f, a1 = 0.0f;
  for (int k = 0; k < T_TEXT; ++k) {
    float p = probs[wib * T_TEXT + k];
    const float* vr = Vt + ((size_t)(b * T_TEXT + k)) * D_MODEL + h * DH;
    a0 += p * vr[d0];
    a1 += p * vr[d0 + 1];
  }
  float* o = AO + ((size_t)(b * TT + q)) * D_MODEL + h * DH;
  o[d0] = a0;
  o[d0 + 1] = a1;
}

// ---------------------------------------------------------------------------
// 8) Style modulation: h = gamma * h + beta (gamma=gb[:, :512], beta=gb[:, 512:])
// ---------------------------------------------------------------------------
__global__ __launch_bounds__(256) void modulate_kernel(
    float* __restrict__ H, const float* __restrict__ GB) {
  int idx = blockIdx.x * 256 + threadIdx.x;        // over 2048*512
  int d = idx & (D_MODEL - 1);
  int r = idx >> 9;
  int b = r >> 10;
  H[idx] = GB[b * 2 * D_MODEL + d] * H[idx] + GB[b * 2 * D_MODEL + D_MODEL + d];
}

// ---------------------------------------------------------------------------
// 9) Zero-pad z_style [2,256] -> [16,256] so the style GEMM needs no predicates
// ---------------------------------------------------------------------------
__global__ __launch_bounds__(256) void pad_style_kernel(
    const float* __restrict__ z, float* __restrict__ Z16) {
  int idx = blockIdx.x * 256 + threadIdx.x;        // 16*256 = 4096
  int r = idx >> 8;
  int c = idx & 255;
  Z16[idx] = (r < BB) ? z[r * D_STYLE + c] : 0.0f;
}

// ---------------------------------------------------------------------------
// Host orchestration
// ---------------------------------------------------------------------------
static inline void launch_gemm(const float* A, int lda, const float* W,
                               const float* bias, const float* R, float* Y,
                               int M, int N, int K, int act, hipStream_t st) {
  int waves = (M >> 4) * (N >> 6);
  int blocks = (waves + 7) >> 3;
  gemm_bf16_kernel<<<blocks, 256, 0, st>>>(A, lda, W, bias, R, Y, M, N, K, act);
}

extern "C" void kernel_launch(void* const* d_in, const int* in_sizes, int n_in,
                              void* d_out, int out_size, void* d_ws, size_t ws_size,
                              hipStream_t stream) {
  (void)in_sizes; (void)n_in; (void)out_size; (void)ws_size;
  // ---- inputs (setup_inputs dict order, params flattened in insertion order)
  const int*   audio_tokens = (const int*)d_in[0];
  const float* text_hidden  = (const float*)d_in[1];
  const float* z_style      = (const float*)d_in[2];
  const int*   text_mask    = (const int*)d_in[3];
  int p = 4;
  const float* tok_emb  = (const float*)d_in[p++];
  const float* pos_emb  = (const float*)d_in[p++];
  const float* quant_emb= (const float*)d_in[p++];
  const float* ln_m_w   = (const float*)d_in[p++];
  const float* ln_m_b   = (const float*)d_in[p++];
  const float* in_w     = (const float*)d_in[p++];
  const float* conv_w   = (const float*)d_in[p++];
  const float* conv_b   = (const float*)d_in[p++];
  const float* xproj_w  = (const float*)d_in[p++];
  const float* dt_w     = (const float*)d_in[p++];
  const float* dt_b     = (const float*)d_in[p++];
  const float* A_log    = (const float*)d_in[p++];
  const float* Dp       = (const float*)d_in[p++];
  const float* out_w    = (const float*)d_in[p++];
  const float* ln_c_w   = (const float*)d_in[p++];
  const float* ln_c_b   = (const float*)d_in[p++];
  const float* Wq       = (const float*)d_in[p++];
  const float* bq       = (const float*)d_in[p++];
  const float* Wk       = (const float*)d_in[p++];
  const float* bk       = (const float*)d_in[p++];
  const float* Wv       = (const float*)d_in[p++];
  const float* bv       = (const float*)d_in[p++];
  const float* Wo       = (const float*)d_in[p++];
  const float* bo       = (const float*)d_in[p++];
  const float* ln_f_w   = (const float*)d_in[p++];
  const float* ln_f_b   = (const float*)d_in[p++];
  const float* style_w  = (const float*)d_in[p++];
  const float* style_b  = (const float*)d_in[p++];
  const float* ff1_w    = (const float*)d_in[p++];
  const float* ff1_b    = (const float*)d_in[p++];
  const float* ff2_w    = (const float*)d_in[p++];
  const float* ff2_b    = (const float*)d_in[p++];
  const float* no_w     = (const float*)d_in[p++];
  const float* no_b     = (const float*)d_in[p++];
  const float* head_w   = (const float*)d_in[p++];
  const float* head_b   = (const float*)d_in[p++];

  // ---- workspace layout (floats), ~49 MB total
  float* ws = (float*)d_ws;
  size_t o = 0;
  float* X    = ws + o; o += (size_t)ROWS * D_MODEL;        // residual stream
  float* XN   = ws + o; o += (size_t)ROWS * D_MODEL;        // LN output
  float* TMP1 = ws + o; o += (size_t)ROWS * 2 * D_INNER;    // xz / ff1 out
  float* TMP2 = ws + o; o += (size_t)ROWS * D_INNER;        // conv xi / gate / attn-out
  float* TMP3 = ws + o; o += (size_t)ROWS * D_INNER;        // dt_pre / Q
  float* TMP4 = ws + o; o += (size_t)ROWS * D_INNER;        // ssm y / K,V
  float* DBC  = ws + o; o += (size_t)ROWS * (DT_RANK + 2 * D_STATE);
  float* GB   = ws + o; o += (size_t)16 * 2 * D_MODEL;      // style gamma/beta (padded M)
  float* ZPAD = ws + o; o += (size_t)16 * D_STYLE;          // zero-padded z_style

  embed_kernel<<<ROWS, 256, 0, stream>>>(audio_tokens, tok_emb, pos_emb,
                                         quant_emb, X);
  pad_style_kernel<<<16, 256, 0, stream>>>(z_style, ZPAD);

  for (int l = 0; l < N_LAYERS; ++l) {
    // ---------------- Mamba block ----------------
    ln_kernel<<<ROWS / 8, 256, 0, stream>>>(X, ln_m_w + l * D_MODEL,
                                            ln_m_b + l * D_MODEL, XN, ROWS);
    launch_gemm(XN, D_MODEL, in_w + (size_t)l * 2 * D_INNER * D_MODEL, nullptr,
                nullptr, TMP1, ROWS, 2 * D_INNER, D_MODEL, 0, stream);
    conv_silu_kernel<<<(ROWS * D_INNER) / 256, 256, 0, stream>>>(
        TMP1, conv_w + (size_t)l * D_INNER * D_CONV, conv_b + l * D_INNER, TMP2);
    launch_gemm(TMP2, D_INNER, xproj_w + (size_t)l * 64 * D_INNER, nullptr,
                nullptr, DBC, ROWS, DT_RANK + 2 * D_STATE, D_INNER, 0, stream);
    launch_gemm(DBC, DT_RANK + 2 * D_STATE, dt_w + (size_t)l * D_INNER * DT_RANK,
                dt_b + l * D_INNER, nullptr, TMP3, ROWS, D_INNER, DT_RANK, 0,
                stream);
    scan_kernel<<<(BB * D_INNER) / 256, 256, 0, stream>>>(
        TMP3, TMP2, DBC, A_log + (size_t)l * D_INNER * D_STATE,
        Dp + l * D_INNER, TMP4);
    gate_kernel<<<(ROWS * D_INNER) / 256, 256, 0, stream>>>(TMP4, TMP1, TMP2);
    launch_gemm(TMP2, D_INNER, out_w + (size_t)l * D_MODEL * D_INNER, nullptr,
                X, X, ROWS, D_MODEL, D_INNER, 0, stream);   // residual fused

    // ---------------- Cross-attention ----------------
    ln_kernel<<<ROWS / 8, 256, 0, stream>>>(X, ln_c_w + l * D_MODEL,
                                            ln_c_b + l * D_MODEL, XN, ROWS);
    launch_gemm(XN, D_MODEL, Wq + (size_t)l * D_MODEL * D_MODEL,
                bq + l * D_MODEL, nullptr, TMP3, ROWS, D_MODEL, D_MODEL, 0,
                stream);
    launch_gemm(text_hidden, D_MODEL, Wk + (size_t)l * D_MODEL * D_MODEL,
                bk + l * D_MODEL, nullptr, TMP4, KV_ROWS, D_MODEL, D_MODEL, 0,
                stream);
    launch_gemm(text_hidden, D_MODEL, Wv + (size_t)l * D_MODEL * D_MODEL,
                bv + l * D_MODEL, nullptr, TMP4 + (size_t)KV_ROWS * D_MODEL,
                KV_ROWS, D_MODEL, D_MODEL, 0, stream);
    attn_kernel<<<(BB * N_HEADS * TT) / 8, 256, 0, stream>>>(
        TMP3, TMP4, TMP4 + (size_t)KV_ROWS * D_MODEL, text_mask, TMP2);
    launch_gemm(TMP2, D_MODEL, Wo + (size_t)l * D_MODEL * D_MODEL,
                bo + l * D_MODEL, X, X, ROWS, D_MODEL, D_MODEL, 0, stream);

    // ---------------- Style-modulated FFN ----------------
    ln_kernel<<<ROWS / 8, 256, 0, stream>>>(X, ln_f_w + l * D_MODEL,
                                            ln_f_b + l * D_MODEL, XN, ROWS);
    launch_gemm(ZPAD, D_STYLE, style_w + (size_t)l * 2 * D_MODEL * D_STYLE,
                style_b + l * 2 * D_MODEL, nullptr, GB, 16, 2 * D_MODEL,
                D_STYLE, 2 /*tanh*/, stream);
    modulate_kernel<<<(ROWS * D_MODEL) / 256, 256, 0, stream>>>(XN, GB);
    launch_gemm(XN, D_MODEL, ff1_w + (size_t)l * D_FF * D_MODEL,
                ff1_b + l * D_FF, nullptr, TMP1, ROWS, D_FF, D_MODEL,
                1 /*gelu*/, stream);
    launch_gemm(TMP1, D_FF, ff2_w + (size_t)l * D_MODEL * D_FF,
                ff2_b + l * D_MODEL, X, X, ROWS, D_MODEL, D_FF, 0, stream);
  }

  // ---------------- Head ----------------
  ln_kernel<<<ROWS / 8, 256, 0, stream>>>(X, no_w, no_b, XN, ROWS);
  launch_gemm(XN, D_MODEL, head_w, head_b, nullptr, (float*)d_out, ROWS, VOCAB,
              D_MODEL, 0, stream);
}
